// PureSparseBackboneCol_Res1_38079180046734
// MI455X (gfx1250) — matrix operand order repaired and checked by reading, not compiled, not run on hardware
//
#include <hip/hip_runtime.h>

typedef __bf16 bf16;
typedef __attribute__((ext_vector_type(16))) __bf16 v16bf;
typedef __attribute__((ext_vector_type(8)))  __bf16 v8bf;
typedef __attribute__((ext_vector_type(8)))  float  v8f;

#define CDIV(a, b) (((a) + (b) - 1) / (b))

// ------------------------------------------------------------------
// utility kernels
// ------------------------------------------------------------------
__global__ void k_zero_f32(float* p, long n) {
  long t = (long)blockIdx.x * blockDim.x + threadIdx.x;
  if (t < n) p[t] = 0.f;
}
__global__ void k_zero_bf16(bf16* p, long n) {
  long t = (long)blockIdx.x * blockDim.x + threadIdx.x;
  if (t < n) p[t] = (bf16)0.f;
}

// concat [x_geo | x_col] -> (n_vox+1, 4) f32, pad row zero
__global__ void k_concat(const float* __restrict__ geo, const float* __restrict__ col,
                         float* __restrict__ out, int n_vox) {
  long t = (long)blockIdx.x * blockDim.x + threadIdx.x;
  long total = (long)(n_vox + 1) * 4;
  if (t >= total) return;
  int row = (int)(t >> 2), c = (int)(t & 3);
  float v = 0.f;
  if (row < n_vox) v = (c == 0) ? geo[row] : col[(long)row * 3 + c - 1];
  out[t] = v;
}

// Pack fp32 weights (S, CIN, COUT) row-major into bf16 WMMA B-fragment layout:
// [s][kt][nt][lane(32)][e(16)], lane: N = nt*16 + lane%16, K = kt*32 + (lane>=16)*8 + (e&7) + (e>=8)*16
__global__ void k_pack_w(const float* __restrict__ W, bf16* __restrict__ out,
                         int S, int CIN, int COUT) {
  long t = (long)blockIdx.x * blockDim.x + threadIdx.x;
  long total = (long)S * CIN * COUT;
  if (t >= total) return;
  int e = (int)(t & 15);  long r = t >> 4;
  int lane = (int)(r & 31); r >>= 5;
  int NT = COUT >> 4;
  int nt = (int)(r % NT); r /= NT;
  int KT = CIN >> 5;
  int kt = (int)(r % KT); r /= KT;
  int s  = (int)r;
  int n  = nt * 16 + (lane & 15);
  int K  = kt * 32 + ((lane >> 4) << 3) + (e & 7) + ((e >> 3) << 4);
  out[t] = (bf16)W[((long)s * CIN + K) * COUT + n];
}

// conv0: k5 s2, 4 -> 32 channels, 125 offsets, scalar VALU (small FLOP share), +ReLU
__global__ void k_conv0(const float* __restrict__ f0,   // (n_vox+1, 4), pad row zero
                        const float* __restrict__ w0,   // (125, 4, 32)
                        const int*   __restrict__ imap, // (125, n_out), sentinel n_vox
                        int n_out,
                        float* __restrict__ xf,         // (n_out, 32)
                        bf16*  __restrict__ xb) {       // (n_out+1, 32)
  long t = (long)blockIdx.x * blockDim.x + threadIdx.x;
  long total = (long)n_out * 32;
  if (t >= total) return;
  int j = (int)(t >> 5), c = (int)(t & 31);
  float acc = 0.f;
  for (int k = 0; k < 125; ++k) {
    int idx = imap[(long)k * n_out + j];
    const float* fr = f0 + (long)idx * 4;
    const float* wr = w0 + (long)(k * 4) * 32 + c;
    acc = fmaf(fr[0], wr[0], acc);
    acc = fmaf(fr[1], wr[32], acc);
    acc = fmaf(fr[2], wr[64], acc);
    acc = fmaf(fr[3], wr[96], acc);
  }
  acc = fmaxf(acc, 0.f);
  xf[t] = acc;
  xb[t] = (bf16)acc;
}

// ------------------------------------------------------------------
// Core sparse conv: implicit GEMM over 27 offsets using v_wmma_f32_16x16x32_bf16.
// Each wave owns MT 16-row sub-tiles (B fragments shared across sub-tiles).
// fin has a zero row at index n_in (sentinel) -> branch-free gather.
// ------------------------------------------------------------------
union AFrag { v16bf v; v8bf h[2]; };

template <int CIN, int COUT, int MT>
__global__ __launch_bounds__(256)
void k_sparse_conv_wmma(const bf16* __restrict__ fin,     // (n_in+1, CIN) bf16
                        const bf16* __restrict__ pw,      // packed [27][KT][NT][32][16]
                        const int*  __restrict__ in_map,  // (27, n_out)
                        int n_out,
                        float* __restrict__ out_f32,      // (n_out, COUT)
                        bf16*  __restrict__ out_b16) {    // optional (n_out+1, COUT)
  constexpr int KT = CIN / 32, NT = COUT / 16;
  const int lane = threadIdx.x & 31;
  const int wave = threadIdx.x >> 5;
  const int tile = blockIdx.x * (blockDim.x >> 5) + wave;
  const int j0 = tile * (16 * MT);
  if (j0 >= n_out) return;                 // wave-uniform exit (EXEC stays all-ones)
  const int m  = lane & 15;
  const int hi = lane >> 4;

  // clamped gather rows for each sub-tile (invalid rows recompute garbage, masked at store)
  int jc[MT];
#pragma unroll
  for (int mt = 0; mt < MT; ++mt) {
    int j = j0 + mt * 16 + m;
    jc[mt] = (j < n_out) ? j : (n_out - 1);
  }

  v8f acc[MT][NT] = {};

  // software pipeline: index for offset k+1 loaded during offset k
  int idx[MT];
#pragma unroll
  for (int mt = 0; mt < MT; ++mt) idx[mt] = in_map[jc[mt]];

  for (int k = 0; k < 27; ++k) {
    int idx_next[MT];
#pragma unroll
    for (int mt = 0; mt < MT; ++mt) {
      idx_next[mt] = (k < 26) ? in_map[(long)(k + 1) * n_out + jc[mt]] : idx[mt];
      __builtin_prefetch(fin + (long)idx_next[mt] * CIN, 0, 1);  // global_prefetch
    }

#pragma unroll
    for (int kt = 0; kt < KT; ++kt) {
      // 16-bit A layout: lanes<16 hold K {0..7,16..23}, lanes>=16 hold K {8..15,24..31}
      AFrag a[MT];
#pragma unroll
      for (int mt = 0; mt < MT; ++mt) {
        const bf16* p0 = fin + (long)idx[mt] * CIN + kt * 32 + hi * 8;
        a[mt].h[0] = *(const v8bf*)p0;
        a[mt].h[1] = *(const v8bf*)(p0 + 16);
      }
#pragma unroll
      for (int nt = 0; nt < NT; ++nt) {
        const v16bf b = *(const v16bf*)(pw + (((long)(k * KT + kt) * NT + nt) * 32 + lane) * 16);
#pragma unroll
        for (int mt = 0; mt < MT; ++mt) {
          acc[mt][nt] = __builtin_amdgcn_wmma_f32_16x16x32_bf16(
              false, a[mt].v, false, b, (short)0, acc[mt][nt], false, false);
        }
      }
    }
#pragma unroll
    for (int mt = 0; mt < MT; ++mt) idx[mt] = idx_next[mt];
  }

  // C/D layout: VGPR v -> row M = v + hi*8, col N = lane%16
  const bool wb = (out_b16 != nullptr);
#pragma unroll
  for (int mt = 0; mt < MT; ++mt) {
#pragma unroll
    for (int nt = 0; nt < NT; ++nt) {
#pragma unroll
      for (int v = 0; v < 8; ++v) {
        const int r = j0 + mt * 16 + v + hi * 8;
        if (r < n_out) {
          const long o = (long)r * COUT + nt * 16 + m;
          const float y = acc[mt][nt][v];
          out_f32[o] = y;
          if (wb) out_b16[o] = (bf16)y;
        }
      }
    }
  }
}

// ------------------------------------------------------------------
// BatchNorm: stats (sum, sumsq per channel) then fused apply
// stats slice layout: [0..C) = sum, [64..64+C) = sumsq  (zeroed beforehand)
// ------------------------------------------------------------------
__global__ void k_bn_stats(const float* __restrict__ x, long total, int C,
                           float* __restrict__ st) {
  const int tid = threadIdx.x;
  const int c = tid & (C - 1);
  float s = 0.f, s2 = 0.f;
  const long stride = (long)gridDim.x * blockDim.x;   // multiple of C (C | 256)
  for (long i = (long)blockIdx.x * blockDim.x + tid; i < total; i += stride) {
    float v = x[i]; s += v; s2 += v * v;
  }
  __shared__ float sh[512];
  sh[tid] = s; sh[256 + tid] = s2;
  __syncthreads();
  for (int off = 128; off >= C; off >>= 1) {
    if (tid < off) { sh[tid] += sh[tid + off]; sh[256 + tid] += sh[256 + tid + off]; }
    __syncthreads();
  }
  if (tid < C) { atomicAdd(&st[c], sh[tid]); atomicAdd(&st[64 + c], sh[256 + tid]); }
}

__global__ void k_bn_apply(const float* __restrict__ x, const float* __restrict__ st,
                           const float* __restrict__ g, const float* __restrict__ b,
                           const float* __restrict__ res,  // nullable residual
                           long n, int C,
                           float* __restrict__ outf,       // nullable f32 out
                           bf16*  __restrict__ outb) {     // bf16 gather buf
  long t = (long)blockIdx.x * blockDim.x + threadIdx.x;
  long total = n * C;
  if (t >= total) return;
  int c = (int)(t % C);
  float inv_n = 1.f / (float)n;
  float mu  = st[c] * inv_n;
  float var = st[64 + c] * inv_n - mu * mu;
  float y = (x[t] - mu) * rsqrtf(var + 1e-5f) * g[c] + b[c];
  if (res)  y += res[t];
  y = fmaxf(y, 0.f);
  if (outf) outf[t] = y;
  outb[t] = (bf16)y;
}

// ------------------------------------------------------------------
extern "C" void kernel_launch(void* const* d_in, const int* in_sizes, int n_in,
                              void* d_out, int out_size, void* d_ws, size_t ws_size,
                              hipStream_t stream) {
  const float* x_geo = (const float*)d_in[0];
  const float* x_col = (const float*)d_in[1];
  const float* w0    = (const float*)d_in[2];
  const float* w_e1  = (const float*)d_in[3];
  const float* g_e1  = (const float*)d_in[4];
  const float* b_e1  = (const float*)d_in[5];
  const float* w2    = (const float*)d_in[6];
  const float* w_e2  = (const float*)d_in[7];
  const float* g_e2  = (const float*)d_in[8];
  const float* b_e2  = (const float*)d_in[9];
  const int* m1_in = (const int*)d_in[10];
  const int* m2_in = (const int*)d_in[12];
  const int* m3_in = (const int*)d_in[14];
  const int* m4_in = (const int*)d_in[16];

  const int n_vox = in_sizes[0];
  const int n_c2  = in_sizes[10] / 125;
  const int n_c4  = in_sizes[14] / 27;
  (void)n_in; (void)out_size; (void)ws_size;

  // workspace carve (256B aligned slabs)
  char* p = (char*)d_ws;
  auto carve = [&](size_t bytes) {
    char* r = p; p += (bytes + 255) & ~(size_t)255; return (void*)r;
  };
  float* feats0 = (float*)carve((size_t)(n_vox + 1) * 4 * 4);
  float* xf_c2  = (float*)carve((size_t)n_c2 * 32 * 4);
  float* cf_c2  = (float*)carve((size_t)n_c2 * 32 * 4);
  bf16*  xb1_c2 = (bf16*) carve((size_t)(n_c2 + 1) * 32 * 2);
  bf16*  xb2_c2 = (bf16*) carve((size_t)(n_c2 + 1) * 32 * 2);
  float* xf_c4  = (float*)carve((size_t)n_c4 * 64 * 4);
  float* cf_c4  = (float*)carve((size_t)n_c4 * 64 * 4);
  bf16*  xb1_c4 = (bf16*) carve((size_t)(n_c4 + 1) * 64 * 2);
  bf16*  xb2_c4 = (bf16*) carve((size_t)(n_c4 + 1) * 64 * 2);
  float* stats  = (float*)carve((size_t)8 * 128 * 4);
  bf16*  pw_e1  = (bf16*) carve((size_t)108 * 1 * 2 * 32 * 16 * 2);
  bf16*  pw2    = (bf16*) carve((size_t)27  * 1 * 4 * 32 * 16 * 2);
  bf16*  pw_e2  = (bf16*) carve((size_t)108 * 2 * 4 * 32 * 16 * 2);

  // ---- weight packing (fp32 -> bf16 WMMA-B layout) ----
  { long t = 108L * 32 * 32; k_pack_w<<<CDIV(t, 256), 256, 0, stream>>>(w_e1, pw_e1, 108, 32, 32); }
  { long t = 27L  * 32 * 64; k_pack_w<<<CDIV(t, 256), 256, 0, stream>>>(w2,   pw2,   27,  32, 64); }
  { long t = 108L * 64 * 64; k_pack_w<<<CDIV(t, 256), 256, 0, stream>>>(w_e2, pw_e2, 108, 64, 64); }

  // ---- inputs + zero pad rows / stats ----
  { long t = (long)(n_vox + 1) * 4; k_concat<<<CDIV(t, 256), 256, 0, stream>>>(x_geo, x_col, feats0, n_vox); }
  k_zero_f32<<<CDIV(8L * 128, 256), 256, 0, stream>>>(stats, 8 * 128);
  k_zero_bf16<<<1, 64, 0, stream>>>(xb1_c2 + (long)n_c2 * 32, 32);
  k_zero_bf16<<<1, 64, 0, stream>>>(xb2_c2 + (long)n_c2 * 32, 32);
  k_zero_bf16<<<1, 64, 0, stream>>>(xb1_c4 + (long)n_c4 * 64, 64);
  k_zero_bf16<<<1, 64, 0, stream>>>(xb2_c4 + (long)n_c4 * 64, 64);

  // ---- conv0 (k5 s2, 4->32) + ReLU ----
  { long t = (long)n_c2 * 32; k_conv0<<<CDIV(t, 256), 256, 0, stream>>>(feats0, w0, m1_in, n_c2, xf_c2, xb1_c2); }

  const int blocks_c2 = CDIV(CDIV(n_c2, 32), 8);   // MT=2 at c2 level
  const int blocks_c4 = CDIV(CDIV(n_c4, 16), 8);   // MT=1 at c4 level
  long ec2 = (long)n_c2 * 32, ec4 = (long)n_c4 * 64;
  int sblk2 = (int)((ec2 + 255) / 256); if (sblk2 > 1024) sblk2 = 1024;
  int sblk4 = (int)((ec4 + 255) / 256); if (sblk4 > 1024) sblk4 = 1024;

  // ---- enc1: 2 BasicBlocks at c2 (32 ch) ----
  for (int i = 0; i < 2; ++i) {
    const size_t wsl = (size_t)27 * 1 * 2 * 32 * 16;  // elems per (block,conv) slice
    const bf16* pwA = pw_e1 + (size_t)(i * 2 + 0) * wsl;
    const bf16* pwB = pw_e1 + (size_t)(i * 2 + 1) * wsl;
    const float* gA = g_e1 + (i * 2 + 0) * 32, *bA = b_e1 + (i * 2 + 0) * 32;
    const float* gB = g_e1 + (i * 2 + 1) * 32, *bB = b_e1 + (i * 2 + 1) * 32;
    float* stA = stats + (i * 2 + 0) * 128;
    float* stB = stats + (i * 2 + 1) * 128;

    k_sparse_conv_wmma<32, 32, 2><<<blocks_c2, 256, 0, stream>>>(xb1_c2, pwA, m2_in, n_c2, cf_c2, (bf16*)nullptr);
    k_bn_stats<<<sblk2, 256, 0, stream>>>(cf_c2, ec2, 32, stA);
    k_bn_apply<<<CDIV(ec2, 256), 256, 0, stream>>>(cf_c2, stA, gA, bA, (const float*)nullptr,
                                                   n_c2, 32, (float*)nullptr, xb2_c2);
    k_sparse_conv_wmma<32, 32, 2><<<blocks_c2, 256, 0, stream>>>(xb2_c2, pwB, m2_in, n_c2, cf_c2, (bf16*)nullptr);
    k_bn_stats<<<sblk2, 256, 0, stream>>>(cf_c2, ec2, 32, stB);
    k_bn_apply<<<CDIV(ec2, 256), 256, 0, stream>>>(cf_c2, stB, gB, bB, xf_c2,
                                                   n_c2, 32, xf_c2, xb1_c2);
  }

  // ---- conv2 (32->64 onto c4 coords), no BN/ReLU ----
  k_sparse_conv_wmma<32, 64, 1><<<blocks_c4, 256, 0, stream>>>(xb1_c2, pw2, m3_in, n_c4, xf_c4, xb1_c4);

  // ---- enc2: 2 BasicBlocks at c4 (64 ch) ----
  for (int i = 0; i < 2; ++i) {
    const size_t wsl = (size_t)27 * 2 * 4 * 32 * 16;
    const bf16* pwA = pw_e2 + (size_t)(i * 2 + 0) * wsl;
    const bf16* pwB = pw_e2 + (size_t)(i * 2 + 1) * wsl;
    const float* gA = g_e2 + (i * 2 + 0) * 64, *bA = b_e2 + (i * 2 + 0) * 64;
    const float* gB = g_e2 + (i * 2 + 1) * 64, *bB = b_e2 + (i * 2 + 1) * 64;
    float* stA = stats + (4 + i * 2 + 0) * 128;
    float* stB = stats + (4 + i * 2 + 1) * 128;

    k_sparse_conv_wmma<64, 64, 1><<<blocks_c4, 256, 0, stream>>>(xb1_c4, pwA, m4_in, n_c4, cf_c4, (bf16*)nullptr);
    k_bn_stats<<<sblk4, 256, 0, stream>>>(cf_c4, ec4, 64, stA);
    k_bn_apply<<<CDIV(ec4, 256), 256, 0, stream>>>(cf_c4, stA, gA, bA, (const float*)nullptr,
                                                   n_c4, 64, (float*)nullptr, xb2_c4);
    k_sparse_conv_wmma<64, 64, 1><<<blocks_c4, 256, 0, stream>>>(xb2_c4, pwB, m4_in, n_c4, cf_c4, (bf16*)nullptr);
    k_bn_stats<<<sblk4, 256, 0, stream>>>(cf_c4, ec4, 64, stB);
    float* fout = (i == 1) ? (float*)d_out : xf_c4;
    k_bn_apply<<<CDIV(ec4, 256), 256, 0, stream>>>(cf_c4, stB, gB, bB, xf_c4,
                                                   n_c4, 64, fout, xb1_c4);
  }
}
